// GeoGNN_76965813944555
// MI455X (gfx1250) — matrix operand order repaired and checked by reading, not compiled, not run on hardware
//
#include <hip/hip_runtime.h>
#include <hip/hip_bf16.h>
#include <hip/hip_fp16.h>

typedef _Float16 v16h __attribute__((ext_vector_type(16)));
typedef _Float16 v8h  __attribute__((ext_vector_type(8)));
typedef float    v8f  __attribute__((ext_vector_type(8)));

#define ACT_STRIDE 256
#define WPB 4   // waves per block (128 threads, wave32)

// f16 weight blob offsets (in f16 elements), each Wt stored [Npad][Kpad]
#define OFF_IN    0        // 64 x 32
#define OFF_G0    2048     // 256 x 64
#define OFF_G1    18432    // 256 x 256
#define OFF_G2    83968    // 64 x 256
#define OFF_GC1   100352   // 64 x 64
#define OFF_GC2   104448   // 64 x 64
#define OFF_D1    108544   // 64 x 64
#define OFF_D2    112640   // 32 x 64
#define OFF_D3    114688   // 16 x 32
#define WBLOB_F16 115200

// ---------------- weight prep: f32 [K][Nout] -> f16 transposed [Npad][Kpad] ----------------
__global__ void prep_w_kernel(const float* __restrict__ src, _Float16* __restrict__ dst,
                              int K, int Nout, int Kpad, int Npad)
{
  int tot = Kpad * Npad;
  for (int i = blockIdx.x * blockDim.x + threadIdx.x; i < tot; i += gridDim.x * blockDim.x) {
    int n = i / Kpad, k = i - n * Kpad;
    float v = (k < K && n < Nout) ? src[(size_t)k * Nout + n] : 0.0f;
    dst[(size_t)n * Kpad + k] = (_Float16)v;
  }
}

// ---------------- in-degree indicator ----------------
__global__ void ind_zero_kernel(float* __restrict__ ind, int n)
{
  int i = blockIdx.x * blockDim.x + threadIdx.x;
  if (i < n) ind[i] = 0.0f;
}

__global__ void ind_set_kernel(const int* __restrict__ col, float* __restrict__ ind, int e)
{
  int i = blockIdx.x * blockDim.x + threadIdx.x;
  if (i < e) ind[col[i]] = 1.0f;   // benign races: all write 1.0
}

// ---------------- wave-level WMMA GEMM over a 16-row LDS activation tile ----------------
// A: 16 x (KT*32) f16 rows in LDS (row stride ACT_STRIDE)
// B: weights f16 transposed [nt*16+n][k], contiguous K
// D: NT tiles of 16x16 f32 accumulators
template<int KT, int NT>
__device__ __forceinline__ void wave_gemm(const _Float16* __restrict__ Wt,
                                          const _Float16* act, v8f (&acc)[NT],
                                          int r, int h)
{
#pragma unroll
  for (int nt = 0; nt < NT; ++nt)
#pragma unroll
    for (int i = 0; i < 8; ++i) acc[nt][i] = 0.0f;

#pragma unroll
  for (int kt = 0; kt < KT; ++kt) {
    // A layout (contiguous, matching B): lane half h reads K in [16h,16h+16)
    const _Float16* ab = act + r * ACT_STRIDE + kt * 32;
    v8h lo = *(const v8h*)(ab + 16 * h);
    v8h hi = *(const v8h*)(ab + 16 * h + 8);
    v16h a = __builtin_shufflevector(lo, hi, 0,1,2,3,4,5,6,7,8,9,10,11,12,13,14,15);
#pragma unroll
    for (int nt = 0; nt < NT; ++nt) {
      // B layout (contiguous k, matching A): lane (n = r) holds B[16h .. 16h+15][n] -> contiguous in k
      v16h b = *(const v16h*)(Wt + (size_t)(nt * 16 + r) * (KT * 32) + kt * 32 + 16 * h);
      acc[nt] = __builtin_amdgcn_wmma_f32_16x16x32_f16(false, a, false, b,
                                                       (short)0, acc[nt], false, false);
    }
  }
}

enum { A_NONE = 0, A_RELU = 1, A_ELU = 2 };

// Fused epilogue: bias, activation, indicator gating, residual, optional save copy.
// D layout: lane holds D[m = g + 8h][n = nt*16 + r].
template<int NT, int ACT, bool IND, bool RES, bool HASB, bool SAVE>
__device__ __forceinline__ void epilogue(v8f (&acc)[NT], const float* __restrict__ bias,
                                         _Float16* act, _Float16* sav,
                                         const float (&indv)[8], int r, int h)
{
#pragma unroll
  for (int nt = 0; nt < NT; ++nt) {
    int n = nt * 16 + r;
    float b = HASB ? bias[n] : 0.0f;
#pragma unroll
    for (int g = 0; g < 8; ++g) {
      int m = g + 8 * h;
      float v = acc[nt][g] + b;
      if (ACT == A_RELU) v = fmaxf(v, 0.0f);
      if (ACT == A_ELU)  v = (v > 0.0f) ? v : (__expf(v) - 1.0f);
      if (IND) v *= indv[g];
      if (RES) v += (float)act[m * ACT_STRIDE + n];
      act[m * ACT_STRIDE + n] = (_Float16)v;
      if (SAVE) sav[m * 64 + n] = (_Float16)v;
    }
  }
}

// ---------------- fully fused per-node chain ----------------
__global__ __launch_bounds__(32 * WPB)
void geognn_chain_kernel(const float* __restrict__ x,
                         const float* __restrict__ in_b,  const float* __restrict__ gc_b1,
                         const float* __restrict__ gc_b2, const float* __restrict__ d_b1,
                         const float* __restrict__ d_b2,  const float* __restrict__ d_b3,
                         const float* __restrict__ gc_sw_p,
                         const _Float16* __restrict__ wb, const float* __restrict__ ind,
                         float* __restrict__ out, int nN)
{
  __shared__ __align__(16) _Float16 s_act[WPB][16 * ACT_STRIDE];
  __shared__ __align__(16) _Float16 s_sav[WPB][16 * 64];

  const int lane = threadIdx.x & 31;
  const int wid  = threadIdx.x >> 5;
  const int tile = blockIdx.x * WPB + wid;
  const int r0   = tile * 16;
  if (r0 >= nN) return;

  _Float16* act = s_act[wid];
  _Float16* sav = s_sav[wid];
  const int r = lane & 15, h = lane >> 4;
  const float sw = gc_sw_p[0];

  // warm the largest weight block (g1: 128KB) -> global_prefetch_b8
  {
    const char* p = (const char*)(wb + OFF_G1);
    for (int o = (int)threadIdx.x * 128; o < 131072; o += WPB * 32 * 128)
      __builtin_prefetch(p + o, 0, 1);
  }

  // stage x tile as f16, K padded 16 -> 32
  for (int idx = lane; idx < 16 * 32; idx += 32) {
    int rr = idx >> 5, kk = idx & 31;
    float v = 0.0f;
    if (kk < 16 && (r0 + rr) < nN) v = x[(size_t)(r0 + rr) * 16 + kk];
    act[rr * ACT_STRIDE + kk] = (_Float16)v;
  }

  float indv[8];
#pragma unroll
  for (int g = 0; g < 8; ++g) {
    int m = r0 + g + 8 * h;
    indv[g] = (m < nN) ? ind[m] : 0.0f;
  }

  v8f acc16[16];
  v8f acc4[4];
  v8f acc2[2];
  v8f acc1[1];

  // input projection: 16(pad32) -> 64, + bias
  wave_gemm<1, 4>(wb + OFF_IN, act, acc4, r, h);
  epilogue<4, A_NONE, false, false, true, false>(acc4, in_b, act, sav, indv, r, h);

  // GAT0: 64 -> 256, h1 = ind * elu(p)
  wave_gemm<2, 16>(wb + OFF_G0, act, acc16, r, h);
  epilogue<16, A_ELU, true, false, false, false>(acc16, nullptr, act, sav, indv, r, h);

  // GAT1: 256 -> 256, h2 = h1 + ind * elu(p)
  wave_gemm<8, 16>(wb + OFF_G1, act, acc16, r, h);
  epilogue<16, A_ELU, true, true, false, false>(acc16, nullptr, act, sav, indv, r, h);

  // GAT2: 256 -> 64, h3 = ind * elu(p); save h3 for gc residual
  wave_gemm<8, 4>(wb + OFF_G2, act, acc4, r, h);
  epilogue<4, A_ELU, true, false, false, true>(acc4, nullptr, act, sav, indv, r, h);

  // GC1: 64 -> 64, relu + bias
  wave_gemm<2, 4>(wb + OFF_GC1, act, acc4, r, h);
  epilogue<4, A_RELU, false, false, true, false>(acc4, gc_b1, act, sav, indv, r, h);

  // GC2: 64 -> 64 + bias; h4 = h3 + sw * t
  wave_gemm<2, 4>(wb + OFF_GC2, act, acc4, r, h);
#pragma unroll
  for (int nt = 0; nt < 4; ++nt) {
    int n = nt * 16 + r;
    float b = gc_b2[n];
#pragma unroll
    for (int g = 0; g < 8; ++g) {
      int m = g + 8 * h;
      float t  = acc4[nt][g] + b;
      float hv = (float)sav[m * 64 + n] + sw * t;
      act[m * ACT_STRIDE + n] = (_Float16)hv;
    }
  }

  // D1: 64 -> 64, relu + bias
  wave_gemm<2, 4>(wb + OFF_D1, act, acc4, r, h);
  epilogue<4, A_RELU, false, false, true, false>(acc4, d_b1, act, sav, indv, r, h);

  // D2: 64 -> 32, relu + bias
  wave_gemm<2, 2>(wb + OFF_D2, act, acc2, r, h);
  epilogue<2, A_RELU, false, false, true, false>(acc2, d_b2, act, sav, indv, r, h);

  // D3: 32 -> 3 (Nout padded to 16), + bias, store f32 output
  wave_gemm<1, 1>(wb + OFF_D3, act, acc1, r, h);
  {
    int n = r;
    if (n < 3) {
      float b = d_b3[n];
#pragma unroll
      for (int g = 0; g < 8; ++g) {
        int m = r0 + g + 8 * h;
        if (m < nN) out[(size_t)m * 3 + n] = acc1[0][g] + b;
      }
    }
  }
}

// ---------------- host launcher ----------------
extern "C" void kernel_launch(void* const* d_in, const int* in_sizes, int n_in,
                              void* d_out, int out_size, void* d_ws, size_t ws_size,
                              hipStream_t stream)
{
  const float* x        = (const float*)d_in[0];
  const int*   edge_idx = (const int*)d_in[2];
  const float* in_W  = (const float*)d_in[3];
  const float* in_b  = (const float*)d_in[4];
  const float* g0_W  = (const float*)d_in[7];
  const float* g1_W  = (const float*)d_in[10];
  const float* g2_W  = (const float*)d_in[13];
  const float* gc_sw = (const float*)d_in[16];
  const float* gc_W1 = (const float*)d_in[17];
  const float* gc_b1 = (const float*)d_in[18];
  const float* gc_W2 = (const float*)d_in[19];
  const float* gc_b2 = (const float*)d_in[20];
  const float* d_W1  = (const float*)d_in[21];
  const float* d_b1  = (const float*)d_in[22];
  const float* d_W2  = (const float*)d_in[23];
  const float* d_b2  = (const float*)d_in[24];
  const float* d_W3  = (const float*)d_in[25];
  const float* d_b3  = (const float*)d_in[26];

  const int N = in_sizes[0] / 16;   // x is [N, 16]
  const int E = in_sizes[1] / 4;    // edge_attr is [E, 4]

  _Float16* wb  = (_Float16*)d_ws;
  float*    ind = (float*)((char*)d_ws + (size_t)WBLOB_F16 * 2);  // 230400B, 256B-aligned

  auto prep = [&](const float* src, int off, int K, int Nout, int Kpad, int Npad) {
    int tot = Kpad * Npad;
    int blocks = (tot + 255) / 256; if (blocks > 256) blocks = 256;
    prep_w_kernel<<<dim3(blocks), dim3(256), 0, stream>>>(src, wb + off, K, Nout, Kpad, Npad);
  };
  prep(in_W,  OFF_IN,  16,  64,  32,  64);
  prep(g0_W,  OFF_G0,  64,  256, 64,  256);
  prep(g1_W,  OFF_G1,  256, 256, 256, 256);
  prep(g2_W,  OFF_G2,  256, 64,  256, 64);
  prep(gc_W1, OFF_GC1, 64,  64,  64,  64);
  prep(gc_W2, OFF_GC2, 64,  64,  64,  64);
  prep(d_W1,  OFF_D1,  64,  64,  64,  64);
  prep(d_W2,  OFF_D2,  64,  32,  64,  32);
  prep(d_W3,  OFF_D3,  32,  3,   32,  16);

  ind_zero_kernel<<<(N + 255) / 256, 256, 0, stream>>>(ind, N);
  ind_set_kernel<<<(E + 255) / 256, 256, 0, stream>>>(edge_idx + E, ind, E);

  int tiles  = (N + 15) / 16;
  int blocks = (tiles + WPB - 1) / WPB;
  geognn_chain_kernel<<<blocks, 32 * WPB, 0, stream>>>(
      x, in_b, gc_b1, gc_b2, d_b1, d_b2, d_b3,
      gc_sw, wb, ind, (float*)d_out, N);

  (void)n_in; (void)out_size; (void)ws_size;
}
